// Prototype_Block_83519934038276
// MI455X (gfx1250) — matrix operand order repaired and checked by reading, not compiled, not run on hardware
//
#include <hip/hip_runtime.h>
#include <math.h>

// ---------------------------------------------------------------------------
// Types for CDNA5 WMMA (gfx1250, wave32)
// ---------------------------------------------------------------------------
typedef __attribute__((ext_vector_type(16))) __bf16 v16bf;
typedef __attribute__((ext_vector_type(8)))  float  v8f;
typedef __attribute__((ext_vector_type(4)))  unsigned int u32x4;

union Frag { u32x4 u[2]; v16bf v; };  // 32 bytes = one 16x16x32 bf16 operand/lane

static __device__ __forceinline__ __bf16 f2bf(float f) {
  unsigned u = __builtin_bit_cast(unsigned, f);
  u += 0x7FFFu + ((u >> 16) & 1u);            // round-to-nearest-even
  unsigned short h = (unsigned short)(u >> 16);
  return __builtin_bit_cast(__bf16, h);
}

// ---------------------------------------------------------------------------
// Problem constants
// ---------------------------------------------------------------------------
#define BATCH 16
#define SEQ   1024
#define PROTO 50
#define CDIM  768
#define NHEAD 12
#define HDIM  64
#define HIDDEN 3072
#define ROWS_X (BATCH * SEQ)    // 16384
#define ROWS_P (BATCH * PROTO)  // 800

// Epilogue modes
#define EP_F32            0
#define EP_RELU_BF16      1
#define EP_BF16           2
#define EP_BIAS_F32       3
#define EP_BIAS_GELU_BF16 4
#define EP_BIAS_RES_F32   5

// ---------------------------------------------------------------------------
// Fragment loaders (per-lane b128 loads matching the 16x16x32 bf16 layouts)
// ---------------------------------------------------------------------------
template<int TM>
static __device__ __forceinline__ void load_a_frags(Frag (&a)[TM],
    const __bf16* __restrict__ A, int lda, int m0, int r16, int kh, int k0, int M)
{
#pragma unroll
  for (int i = 0; i < TM; i++) {
    int r = m0 + i * 16 + r16;
    if (r > M - 1) r = M - 1;                       // clamp ragged M
    const __bf16* p = A + (long)r * lda + k0 + 8 * kh;
    a[i].u[0] = *(const u32x4*)p;                    // K = 8h .. 8h+7
    a[i].u[1] = *(const u32x4*)(p + 16);             // K = 16+8h .. 16+8h+7
  }
}

template<int TN>
static __device__ __forceinline__ void load_b_frags(Frag (&b)[TN],
    const __bf16* __restrict__ BT, int ldb, int n0, int r16, int kh, int k0)
{
#pragma unroll
  for (int j = 0; j < TN; j++) {
    const __bf16* p = BT + (long)(n0 + j * 16 + r16) * ldb + k0 + 8 * kh;
    b[j].u[0] = *(const u32x4*)p;                    // K = 8h .. 8h+7
    b[j].u[1] = *(const u32x4*)(p + 16);             // K = 16+8h .. 16+8h+7
  }
}

// ---------------------------------------------------------------------------
// Generic batched bf16 WMMA GEMM:  C[M,N] = A[M,K] @ BT[N,K]^T  (BT row-major
// holds B transposed). 8 waves/block. Wave tile = (16*TM) x (16*TN).
// Block tile = (WM_*16*TM) x (WN_*16*TN).  REQUIRES K % 64 == 0.
// Software-pipelined: fragments double-buffered so a full WMMA group overlaps
// the next K-slab's global loads.
// ---------------------------------------------------------------------------
template<int WM_, int WN_, int TM, int TN, int EPI>
__global__ __launch_bounds__(WM_ * WN_ * 32)
void gemm_wmma_kernel(const __bf16* __restrict__ A,  int lda, long sA0, long sA1,
                      const __bf16* __restrict__ BT, int ldb, long sB0, long sB1,
                      void* __restrict__ Cvoid,      int ldc, long sC0, long sC1,
                      const float* __restrict__ bias,
                      const float* __restrict__ resid,
                      int M, int N, int K, int batchInner)
{
  const int bz = blockIdx.z;
  const int b0 = bz / batchInner;
  const int b1 = bz - b0 * batchInner;
  A  += b0 * sA0 + b1 * sA1;
  BT += b0 * sB0 + b1 * sB1;
  const long coff = b0 * sC0 + b1 * sC1;

  const int wave = threadIdx.x >> 5;
  const int lane = threadIdx.x & 31;
  const int wm = wave % WM_;
  const int wn = wave / WM_;
  const int BM = WM_ * TM * 16;
  const int BN = WN_ * TN * 16;
  const int m0 = blockIdx.y * BM + wm * TM * 16;
  const int n0 = blockIdx.x * BN + wn * TN * 16;
  const int r16 = lane & 15;
  const int kh  = lane >> 4;

  v8f acc[TM][TN];
  const v8f vzero = {0.f, 0.f, 0.f, 0.f, 0.f, 0.f, 0.f, 0.f};
#pragma unroll
  for (int i = 0; i < TM; i++)
#pragma unroll
    for (int j = 0; j < TN; j++) acc[i][j] = vzero;

  Frag a0[TM], b0f[TN], a1[TM], b1f[TN];
  load_a_frags<TM>(a0, A, lda, m0, r16, kh, 0, M);
  load_b_frags<TN>(b0f, BT, ldb, n0, r16, kh, 0);

  for (int k0 = 0; k0 < K; k0 += 64) {
    // stage k0+32 while the k0 WMMA group runs
    load_a_frags<TM>(a1, A, lda, m0, r16, kh, k0 + 32, M);
    load_b_frags<TN>(b1f, BT, ldb, n0, r16, kh, k0 + 32);
    if (k0 + 64 < K) {  // pull the slab after next toward the near caches
      __builtin_prefetch((const void*)(A  + (long)(m0 + r16) * lda + k0 + 64), 0, 3);
      __builtin_prefetch((const void*)(BT + (long)(n0 + r16) * ldb + k0 + 64), 0, 3);
    }
#pragma unroll
    for (int i = 0; i < TM; i++)
#pragma unroll
      for (int j = 0; j < TN; j++)
        acc[i][j] = __builtin_amdgcn_wmma_f32_16x16x32_bf16(
            false, a0[i].v, false, b0f[j].v, (short)0, acc[i][j], false, false);

    if (k0 + 64 < K) {  // stage k0+64 while the k0+32 WMMA group runs
      load_a_frags<TM>(a0, A, lda, m0, r16, kh, k0 + 64, M);
      load_b_frags<TN>(b0f, BT, ldb, n0, r16, kh, k0 + 64);
    }
#pragma unroll
    for (int i = 0; i < TM; i++)
#pragma unroll
      for (int j = 0; j < TN; j++)
        acc[i][j] = __builtin_amdgcn_wmma_f32_16x16x32_bf16(
            false, a1[i].v, false, b1f[j].v, (short)0, acc[i][j], false, false);
  }

  float*  Cf = (float*)Cvoid;
  __bf16* Cb = (__bf16*)Cvoid;
#pragma unroll
  for (int i = 0; i < TM; i++)
#pragma unroll
    for (int j = 0; j < TN; j++)
#pragma unroll
      for (int v = 0; v < 8; v++) {
        const int gm = m0 + i * 16 + kh * 8 + v;   // C/D layout: VGPR v -> M
        const int gn = n0 + j * 16 + r16;          // lane&15   -> N
        if (gm >= M) continue;
        float x = acc[i][j][v];
        const long off = coff + (long)gm * ldc + gn;
        if (EPI == EP_F32) {
          Cf[off] = x;
        } else if (EPI == EP_RELU_BF16) {
          Cb[off] = f2bf(fmaxf(x, 0.0f));
        } else if (EPI == EP_BF16) {
          Cb[off] = f2bf(x);
        } else if (EPI == EP_BIAS_F32) {
          Cf[off] = x + bias[gn];
        } else if (EPI == EP_BIAS_GELU_BF16) {
          x += bias[gn];
          Cb[off] = f2bf(0.5f * x * (1.0f + erff(x * 0.70710678118654752f)));
        } else {  // EP_BIAS_RES_F32
          Cf[off] = x + bias[gn] + resid[off];
        }
      }
}

// ---------------------------------------------------------------------------
// Weight convert + transpose: W[K][N] fp32 -> WT[N][K] bf16
// ---------------------------------------------------------------------------
__global__ void wcvt_kernel(const float* __restrict__ W, __bf16* __restrict__ WT,
                            int K, int N)
{
  long idx = (long)blockIdx.x * 256 + threadIdx.x;
  if (idx >= (long)K * N) return;
  int k = (int)(idx / N);
  int n = (int)(idx - (long)k * N);
  WT[(long)n * K + k] = f2bf(W[idx]);
}

// ---------------------------------------------------------------------------
// LayerNorm over C=768, fp32 in -> bf16 out. One 256-thread block per row.
// ---------------------------------------------------------------------------
__global__ __launch_bounds__(256)
void ln_kernel(const float* __restrict__ X, const float* __restrict__ g,
               const float* __restrict__ bta, __bf16* __restrict__ out)
{
  const int row = blockIdx.x;
  const int t = threadIdx.x;
  const float* xr = X + (long)row * CDIM;
  __shared__ float red[256];

  float v[3];
  float s = 0.f;
#pragma unroll
  for (int i = 0; i < 3; i++) { v[i] = xr[t + 256 * i]; s += v[i]; }
  red[t] = s; __syncthreads();
  for (int o = 128; o > 0; o >>= 1) { if (t < o) red[t] += red[t + o]; __syncthreads(); }
  const float mean = red[0] * (1.0f / (float)CDIM);
  __syncthreads();

  float s2 = 0.f;
#pragma unroll
  for (int i = 0; i < 3; i++) { float d = v[i] - mean; s2 += d * d; }
  red[t] = s2; __syncthreads();
  for (int o = 128; o > 0; o >>= 1) { if (t < o) red[t] += red[t + o]; __syncthreads(); }
  const float rstd = rsqrtf(red[0] * (1.0f / (float)CDIM) + 1e-5f);

#pragma unroll
  for (int i = 0; i < 3; i++) {
    int c = t + 256 * i;
    out[(long)row * CDIM + c] = f2bf((v[i] - mean) * rstd * g[c] + bta[c]);
  }
}

// ---------------------------------------------------------------------------
// Per-head L2 normalization of q: QF32[B*N,768] -> QB[B,H,N,64] bf16.
// One wave32 per (b,n,h) 64-vector.
// ---------------------------------------------------------------------------
__global__ __launch_bounds__(256)
void l2q_kernel(const float* __restrict__ QF, __bf16* __restrict__ QB)
{
  const int gw = (int)((blockIdx.x * 256 + threadIdx.x) >> 5);
  const int lane = threadIdx.x & 31;
  const int h = gw % NHEAD;
  const int bn = gw / NHEAD;          // b*SEQ + n
  const int n = bn % SEQ;
  const int b = bn / SEQ;
  const float* src = QF + (long)bn * CDIM + h * HDIM;
  float v0 = src[lane], v1 = src[lane + 32];
  float s = v0 * v0 + v1 * v1;
  for (int o = 16; o > 0; o >>= 1) s += __shfl_xor(s, o, 32);
  const float d = fmaxf(sqrtf(s), 1e-12f);
  __bf16* dst = QB + (((long)(b * NHEAD + h)) * SEQ + n) * HDIM;
  dst[lane]      = f2bf(v0 / d);
  dst[lane + 32] = f2bf(v1 / d);
}

// ---------------------------------------------------------------------------
// K/V prep: KVF32[B*P,1536] -> KB[B,H,64,64] (l2norm * learn_scale, zero-pad)
//                            VT[B,H,64(d),64(p)] (transposed V, zero-pad)
// One wave32 per (b,h,p_pad,which).
// ---------------------------------------------------------------------------
__global__ __launch_bounds__(256)
void kvprep_kernel(const float* __restrict__ KVF, const float* __restrict__ scale,
                   __bf16* __restrict__ KB, __bf16* __restrict__ VT)
{
  const int gw = (int)((blockIdx.x * 256 + threadIdx.x) >> 5);
  const int lane = threadIdx.x & 31;
  const int which = gw & 1;
  int t = gw >> 1;
  const int p = t % 64; t /= 64;
  const int h = t % NHEAD;
  const int b = t / NHEAD;

  float v0 = 0.f, v1 = 0.f;
  if (p < PROTO) {
    const float* src = KVF + ((long)(b * PROTO + p)) * (2 * CDIM) + which * CDIM + h * HDIM;
    v0 = src[lane]; v1 = src[lane + 32];
  }
  if (which == 0) {  // k: l2norm then fold learn_scale
    float s = v0 * v0 + v1 * v1;
    for (int o = 16; o > 0; o >>= 1) s += __shfl_xor(s, o, 32);
    const float d = fmaxf(sqrtf(s), 1e-12f);
    const float sc = scale[h];
    __bf16* dst = KB + (((long)(b * NHEAD + h)) * 64 + p) * HDIM;
    dst[lane]      = f2bf(v0 / d * sc);
    dst[lane + 32] = f2bf(v1 / d * sc);
  } else {           // v: store transposed [d][p]
    __bf16* dst = VT + ((long)(b * NHEAD + h)) * (64 * HDIM);
    dst[lane * 64 + p]        = f2bf(v0);
    dst[(lane + 32) * 64 + p] = f2bf(v1);
  }
}

// ---------------------------------------------------------------------------
// Host-side launch
// ---------------------------------------------------------------------------
extern "C" void kernel_launch(void* const* d_in, const int* in_sizes, int n_in,
                              void* d_out, int out_size, void* d_ws, size_t ws_size,
                              hipStream_t stream)
{
  const float* x      = (const float*)d_in[0];
  const float* proto  = (const float*)d_in[1];
  const float* ln1_g  = (const float*)d_in[2];
  const float* ln1_b  = (const float*)d_in[3];
  const float* wq     = (const float*)d_in[4];
  const float* wkv    = (const float*)d_in[5];
  const float* lscale = (const float*)d_in[6];
  const float* wproj  = (const float*)d_in[7];
  const float* bproj  = (const float*)d_in[8];
  const float* ln2_g  = (const float*)d_in[9];
  const float* ln2_b  = (const float*)d_in[10];
  const float* w1     = (const float*)d_in[11];
  const float* b1     = (const float*)d_in[12];
  const float* w2     = (const float*)d_in[13];
  const float* b2     = (const float*)d_in[14];
  float* out = (float*)d_out;

  // --- workspace carve-up (all regions rewritten every call) ---
  char* wsb = (char*)d_ws;
  size_t o = 0;
  auto take = [&](size_t bytes) -> void* {
    void* p = wsb + o;
    o = (o + bytes + 255) & ~(size_t)255;
    return p;
  };
  __bf16* WQT  = (__bf16*)take((size_t)CDIM * CDIM * 2);
  __bf16* WKVT = (__bf16*)take((size_t)CDIM * 2 * CDIM * 2);
  __bf16* WPT  = (__bf16*)take((size_t)CDIM * CDIM * 2);
  __bf16* W1T  = (__bf16*)take((size_t)CDIM * HIDDEN * 2);
  __bf16* W2T  = (__bf16*)take((size_t)HIDDEN * CDIM * 2);
  __bf16* XN   = (__bf16*)take((size_t)ROWS_X * CDIM * 2);   // LN1(x); later HN
  __bf16* PN   = (__bf16*)take((size_t)ROWS_P * CDIM * 2);   // LN1(proto)
  float*  QF32 = (float*)take((size_t)ROWS_X * CDIM * 4);    // q f32; later Y
  float*  KVF  = (float*)take((size_t)ROWS_P * 2 * CDIM * 4);
  __bf16* QB   = (__bf16*)take((size_t)ROWS_X * CDIM * 2);   // [B,H,N,64]
  __bf16* KB   = (__bf16*)take((size_t)BATCH * NHEAD * 64 * HDIM * 2);
  __bf16* VT   = (__bf16*)take((size_t)BATCH * NHEAD * 64 * HDIM * 2);
  __bf16* ATT  = (__bf16*)take((size_t)BATCH * NHEAD * SEQ * 64 * 2);
  __bf16* YA   = (__bf16*)take((size_t)ROWS_X * CDIM * 2);
  __bf16* G    = (__bf16*)take((size_t)ROWS_X * HIDDEN * 2);
  __bf16* HN   = XN;     // reuse: XN dead after Q GEMM
  float*  Y    = QF32;   // reuse: QF32 dead after l2q

  const dim3 blk(256);

  // --- 1. weights -> bf16 transposed ---
  wcvt_kernel<<<dim3((CDIM * CDIM + 255) / 256), blk, 0, stream>>>(wq, WQT, CDIM, CDIM);
  wcvt_kernel<<<dim3((CDIM * 2 * CDIM + 255) / 256), blk, 0, stream>>>(wkv, WKVT, CDIM, 2 * CDIM);
  wcvt_kernel<<<dim3((CDIM * CDIM + 255) / 256), blk, 0, stream>>>(wproj, WPT, CDIM, CDIM);
  wcvt_kernel<<<dim3((CDIM * HIDDEN + 255) / 256), blk, 0, stream>>>(w1, W1T, CDIM, HIDDEN);
  wcvt_kernel<<<dim3((HIDDEN * CDIM + 255) / 256), blk, 0, stream>>>(w2, W2T, HIDDEN, CDIM);

  // --- 2. LN1 ---
  ln_kernel<<<dim3(ROWS_X), blk, 0, stream>>>(x, ln1_g, ln1_b, XN);
  ln_kernel<<<dim3(ROWS_P), blk, 0, stream>>>(proto, ln1_g, ln1_b, PN);

  // --- 3. Q = XN @ wq  (f32 out for l2norm) ---
  gemm_wmma_kernel<2, 4, 4, 2, EP_F32><<<dim3(CDIM / 128, ROWS_X / 128, 1), blk, 0, stream>>>(
      XN, CDIM, 0, 0, WQT, CDIM, 0, 0, QF32, CDIM, 0, 0,
      nullptr, nullptr, ROWS_X, CDIM, CDIM, 1);

  // --- 4. KV = PN @ wkv (M=800 ragged) ---
  gemm_wmma_kernel<2, 4, 4, 2, EP_F32><<<dim3(2 * CDIM / 128, (ROWS_P + 127) / 128, 1), blk, 0, stream>>>(
      PN, CDIM, 0, 0, WKVT, CDIM, 0, 0, KVF, 2 * CDIM, 0, 0,
      nullptr, nullptr, ROWS_P, 2 * CDIM, CDIM, 1);

  // --- 5. per-head l2norm q, k-prep (fold learn_scale), v-transpose/pad ---
  l2q_kernel<<<dim3(ROWS_X * NHEAD / 8), blk, 0, stream>>>(QF32, QB);
  kvprep_kernel<<<dim3(BATCH * NHEAD * 64 * 2 / 8), blk, 0, stream>>>(KVF, lscale, KB, VT);

  // --- 6. attn = relu(q @ k^T) per (b,h): M=1024, N=64, K=64, 192 batches ---
  gemm_wmma_kernel<4, 2, 2, 2, EP_RELU_BF16><<<dim3(1, SEQ / 128, BATCH * NHEAD), blk, 0, stream>>>(
      QB, HDIM, (long)SEQ * HDIM, 0,
      KB, HDIM, (long)64 * HDIM, 0,
      ATT, 64, (long)SEQ * 64, 0,
      nullptr, nullptr, SEQ, 64, HDIM, 1);

  // --- 7. y_head = attn @ v -> YA[B,N,C] (strided per-head output) ---
  gemm_wmma_kernel<4, 2, 2, 2, EP_BF16><<<dim3(1, SEQ / 128, BATCH * NHEAD), blk, 0, stream>>>(
      ATT, 64, (long)NHEAD * SEQ * 64, (long)SEQ * 64,
      VT, 64, (long)NHEAD * 64 * HDIM, (long)64 * HDIM,
      YA, CDIM, (long)SEQ * CDIM, (long)HDIM,
      nullptr, nullptr, SEQ, HDIM, 64, NHEAD);

  // --- 8. y = YA @ wproj + bproj (f32, kept for residual + LN2) ---
  gemm_wmma_kernel<2, 4, 4, 2, EP_BIAS_F32><<<dim3(CDIM / 128, ROWS_X / 128, 1), blk, 0, stream>>>(
      YA, CDIM, 0, 0, WPT, CDIM, 0, 0, Y, CDIM, 0, 0,
      bproj, nullptr, ROWS_X, CDIM, CDIM, 1);

  // --- 9. LN2 ---
  ln_kernel<<<dim3(ROWS_X), blk, 0, stream>>>(Y, ln2_g, ln2_b, HN);

  // --- 10. G = gelu(HN @ w1 + b1) ---
  gemm_wmma_kernel<2, 4, 4, 2, EP_BIAS_GELU_BF16><<<dim3(HIDDEN / 128, ROWS_X / 128, 1), blk, 0, stream>>>(
      HN, CDIM, 0, 0, W1T, CDIM, 0, 0, G, HIDDEN, 0, 0,
      b1, nullptr, ROWS_X, HIDDEN, CDIM, 1);

  // --- 11. out = y + G @ w2 + b2 ---
  gemm_wmma_kernel<2, 4, 4, 2, EP_BIAS_RES_F32><<<dim3(CDIM / 128, ROWS_X / 128, 1), blk, 0, stream>>>(
      G, HIDDEN, 0, 0, W2T, HIDDEN, 0, 0, out, CDIM, 0, 0,
      b2, Y, ROWS_X, CDIM, HIDDEN, 1);

  (void)in_sizes; (void)n_in; (void)out_size; (void)ws_size;
}
